// MultiDPHConvHeadAttention_87797721465312
// MI455X (gfx1250) — matrix operand order repaired and checked by reading, not compiled, not run on hardware
//
#include <hip/hip_runtime.h>
#include <stdint.h>

namespace {

constexpr int B_ = 8, N_ = 1024, C_ = 768, H_ = 12, D_ = 64, HD_ = 768;

typedef __attribute__((ext_vector_type(16))) __bf16 bf16x16;
typedef __attribute__((ext_vector_type(8)))  float  f32x8;

union FragU { bf16x16 v; uint4 q[2]; };

__device__ inline __bf16 f2bf(float f) {
  union { float f; uint32_t u; } cv; cv.f = f;
  uint32_t r = cv.u + 0x7FFFu + ((cv.u >> 16) & 1u);
  union { unsigned short s; __bf16 b; } o; o.s = (unsigned short)(r >> 16);
  return o.b;
}

__device__ inline f32x8 zero8() {
  f32x8 z = {0.f, 0.f, 0.f, 0.f, 0.f, 0.f, 0.f, 0.f};
  return z;
}

// A fragment (16x32 bf16, row-major source, ld in elements):
// lane l: row = l&15, K = 16*(l>>4) + {0..15} contiguous
__device__ inline bf16x16 loadA(const __bf16* base, int ld) {
  int l = threadIdx.x & 31;
  int r = l & 15;
  int kb = (l >> 4) << 4;
  const __bf16* p = base + (size_t)r * ld + kb;
  FragU f;
  f.q[0] = *reinterpret_cast<const uint4*>(p);
  f.q[1] = *reinterpret_cast<const uint4*>(p + 8);
  return f.v;
}

// B fragment (32x16 bf16) from B^T stored row-major [ncol][K]:
// lane l: col = l&15, K = 16*(l>>4) + {0..15} contiguous
__device__ inline bf16x16 loadB(const __bf16* baseT, int ld) {
  int l = threadIdx.x & 31;
  int c = l & 15;
  int kb = (l >> 4) << 4;
  const __bf16* p = baseT + (size_t)c * ld + kb;
  FragU f;
  f.q[0] = *reinterpret_cast<const uint4*>(p);
  f.q[1] = *reinterpret_cast<const uint4*>(p + 8);
  return f.v;
}

__device__ inline f32x8 wmma_bf16(bf16x16 a, bf16x16 b, f32x8 c) {
  return __builtin_amdgcn_wmma_f32_16x16x32_bf16(false, a, false, b, (short)0, c,
                                                 false, false);
}

// ---------------- elementwise prep ----------------
__global__ __launch_bounds__(256) void cvt_bf16_kernel(const float* in, __bf16* out, int n) {
  int i = blockIdx.x * blockDim.x + threadIdx.x;
  if (i < n) out[i] = f2bf(in[i]);
}

// w[rows][cols] fp32 -> wT[cols][rows] bf16
__global__ __launch_bounds__(256) void transpose_cvt_kernel(const float* w, __bf16* wT,
                                                            int rows, int cols) {
  int i = blockIdx.x * blockDim.x + threadIdx.x;
  if (i >= rows * cols) return;
  int r = i / cols, c = i % cols;
  wT[(size_t)c * rows + r] = f2bf(w[i]);
}

// ---------------- GEMM: C[MxNn] = A[MxK](bf16) * B (BT[NnxK] bf16) + bias, fp32 out
// One wave computes a 32x64 tile: 2 A-frags x 4 B-frags -> 8 independent WMMAs/k-step.
__global__ __launch_bounds__(32) void gemm_bf16_kernel(const __bf16* __restrict__ A,
                                                       const __bf16* __restrict__ BT,
                                                       const float* __restrict__ bias,
                                                       float* __restrict__ C,
                                                       int M, int Nn, int Kk) {
  int row0 = blockIdx.x * 32;
  int n0   = blockIdx.y * 64;
  f32x8 acc[8] = {zero8(), zero8(), zero8(), zero8(),
                  zero8(), zero8(), zero8(), zero8()};
  const __bf16* A0 = A + (size_t)row0 * Kk;
  const __bf16* A1 = A + (size_t)(row0 + 16) * Kk;
  const __bf16* B0 = BT + (size_t)(n0)      * Kk;
  const __bf16* B1 = BT + (size_t)(n0 + 16) * Kk;
  const __bf16* B2 = BT + (size_t)(n0 + 32) * Kk;
  const __bf16* B3 = BT + (size_t)(n0 + 48) * Kk;
  for (int k0 = 0; k0 < Kk; k0 += 32) {
    __builtin_prefetch(A0 + k0 + 256, 0, 1);
    __builtin_prefetch(A1 + k0 + 256, 0, 1);
    bf16x16 a0 = loadA(A0 + k0, Kk);
    bf16x16 a1 = loadA(A1 + k0, Kk);
    bf16x16 b0 = loadB(B0 + k0, Kk);
    bf16x16 b1 = loadB(B1 + k0, Kk);
    bf16x16 b2 = loadB(B2 + k0, Kk);
    bf16x16 b3 = loadB(B3 + k0, Kk);
    acc[0] = wmma_bf16(a0, b0, acc[0]);
    acc[1] = wmma_bf16(a0, b1, acc[1]);
    acc[2] = wmma_bf16(a0, b2, acc[2]);
    acc[3] = wmma_bf16(a0, b3, acc[3]);
    acc[4] = wmma_bf16(a1, b0, acc[4]);
    acc[5] = wmma_bf16(a1, b1, acc[5]);
    acc[6] = wmma_bf16(a1, b2, acc[6]);
    acc[7] = wmma_bf16(a1, b3, acc[7]);
  }
  int l = threadIdx.x & 31;
  int col = l & 15, rsel = l >> 4;
#pragma unroll
  for (int i = 0; i < 2; ++i) {
    int rowbase = row0 + 16 * i;
#pragma unroll
    for (int t = 0; t < 4; ++t) {
      int n = n0 + 16 * t + col;
      float bb = bias ? bias[n] : 0.0f;
      f32x8 a = acc[i * 4 + t];
#pragma unroll
      for (int j = 0; j < 8; ++j) {
        int m = rowbase + j + 8 * rsel;
        C[(size_t)m * Nn + n] = a[j] + bb;
      }
    }
  }
}

// ---------------- depthwise causal conv over batch axis + layout reorder ----------
// y[b] = w0*x[b-2] + w1*x[b-1] + w2*x[b] + cb   (zero history)
// q,k -> [b,h,n,d] bf16 (q scaled by D^-0.5) ; v -> [b,h,d,n] bf16 (transposed)
__global__ __launch_bounds__(256) void conv_reorder_kernel(
    const float* __restrict__ qf, const float* __restrict__ kf, const float* __restrict__ vf,
    const float* __restrict__ cq_w, const float* __restrict__ cq_b,
    const float* __restrict__ ck_w, const float* __restrict__ ck_b,
    const float* __restrict__ cv_w, const float* __restrict__ cv_b,
    __bf16* __restrict__ qc, __bf16* __restrict__ kc, __bf16* __restrict__ vT) {
  int i = blockIdx.x * blockDim.x + threadIdx.x;
  if (i >= N_ * HD_) return;
  int hd = i % HD_, n = i / HD_;
  int h = hd >> 6, d = hd & 63;
  float qw0 = cq_w[hd * 3], qw1 = cq_w[hd * 3 + 1], qw2 = cq_w[hd * 3 + 2], qb = cq_b[hd];
  float kw0 = ck_w[hd * 3], kw1 = ck_w[hd * 3 + 1], kw2 = ck_w[hd * 3 + 2], kb = ck_b[hd];
  float vw0 = cv_w[hd * 3], vw1 = cv_w[hd * 3 + 1], vw2 = cv_w[hd * 3 + 2], vb = cv_b[hd];
  float qp2 = 0.f, qp1 = 0.f, kp2 = 0.f, kp1 = 0.f, vp2 = 0.f, vp1 = 0.f;
  const float scale = 0.125f;  // D^-0.5, folded into q
  for (int b = 0; b < B_; ++b) {
    size_t src = ((size_t)b * N_ + n) * HD_ + hd;
    float qx = qf[src], kx = kf[src], vx = vf[src];
    float qy = qw0 * qp2 + qw1 * qp1 + qw2 * qx + qb;
    float ky = kw0 * kp2 + kw1 * kp1 + kw2 * kx + kb;
    float vy = vw0 * vp2 + vw1 * vp1 + vw2 * vx + vb;
    size_t dq = (((size_t)b * H_ + h) * N_ + n) * D_ + d;
    qc[dq] = f2bf(qy * scale);
    kc[dq] = f2bf(ky);
    vT[(((size_t)b * H_ + h) * D_ + d) * N_ + n] = f2bf(vy);
    qp2 = qp1; qp1 = qx;
    kp2 = kp1; kp1 = kx;
    vp2 = vp1; vp1 = vx;
  }
}

// ---------------- flash attention: 1 wave per 16 query rows -----------------------
__global__ __launch_bounds__(32) void attn_kernel(const __bf16* __restrict__ qc,
                                                  const __bf16* __restrict__ kc,
                                                  const __bf16* __restrict__ vT,
                                                  __bf16* __restrict__ attnO) {
  int n0 = blockIdx.x * 16;
  int bh = blockIdx.y;
  int b = bh / H_, h = bh % H_;
  const __bf16* qb = qc + (size_t)bh * N_ * D_;
  const __bf16* kb = kc + (size_t)bh * N_ * D_;
  const __bf16* vb = vT + (size_t)bh * D_ * N_;
  int l = threadIdx.x & 31;
  int col = l & 15, rsel = l >> 4;

  bf16x16 qA0 = loadA(qb + (size_t)n0 * D_, D_);
  bf16x16 qA1 = loadA(qb + (size_t)n0 * D_ + 32, D_);

  f32x8 o0 = zero8(), o1 = zero8(), o2 = zero8(), o3 = zero8();
  float rm[8], rs[8];
#pragma unroll
  for (int j = 0; j < 8; ++j) { rm[j] = -1e30f; rs[j] = 0.f; }

  __shared__ __align__(16) __bf16 lds[16 * 32];

  for (int m0 = 0; m0 < N_; m0 += 32) {
    // scores: s[16 queries x 32 keys], K-dim = 64 via two chained K32 WMMAs
    f32x8 s0 = zero8(), s1 = zero8();
    bf16x16 kB00 = loadB(kb + (size_t)m0 * D_, D_);
    bf16x16 kB01 = loadB(kb + (size_t)m0 * D_ + 32, D_);
    bf16x16 kB10 = loadB(kb + (size_t)(m0 + 16) * D_, D_);
    bf16x16 kB11 = loadB(kb + (size_t)(m0 + 16) * D_ + 32, D_);
    s0 = wmma_bf16(qA0, kB00, s0);
    s1 = wmma_bf16(qA0, kB10, s1);
    s0 = wmma_bf16(qA1, kB01, s0);
    s1 = wmma_bf16(qA1, kB11, s1);

    // online softmax; rows of a C fragment live in 16-lane halves -> xor 1/2/4/8
#pragma unroll
    for (int j = 0; j < 8; ++j) {
      float t = fmaxf(s0[j], s1[j]);
      t = fmaxf(t, __shfl_xor(t, 1, 32));
      t = fmaxf(t, __shfl_xor(t, 2, 32));
      t = fmaxf(t, __shfl_xor(t, 4, 32));
      t = fmaxf(t, __shfl_xor(t, 8, 32));
      float mnew  = fmaxf(rm[j], t);
      float alpha = __expf(rm[j] - mnew);
      float p0 = __expf(s0[j] - mnew);
      float p1 = __expf(s1[j] - mnew);
      float rsum = p0 + p1;
      rsum += __shfl_xor(rsum, 1, 32);
      rsum += __shfl_xor(rsum, 2, 32);
      rsum += __shfl_xor(rsum, 4, 32);
      rsum += __shfl_xor(rsum, 8, 32);
      rs[j] = rs[j] * alpha + rsum;
      rm[j] = mnew;
      o0[j] *= alpha; o1[j] *= alpha; o2[j] *= alpha; o3[j] *= alpha;
      int row = j + 8 * rsel;
      lds[row * 32 + col]      = f2bf(p0);   // C layout -> LDS (row-major 16x32)
      lds[row * 32 + col + 16] = f2bf(p1);
    }
    __syncthreads();
    bf16x16 pA = loadA(lds, 32);             // re-read in A layout
    __syncthreads();

    // o += P(16x32) @ V(32x64); vT[d][m] is exactly B^T for loadB
    bf16x16 vB0 = loadB(vb + m0, N_);
    bf16x16 vB1 = loadB(vb + (size_t)16 * N_ + m0, N_);
    bf16x16 vB2 = loadB(vb + (size_t)32 * N_ + m0, N_);
    bf16x16 vB3 = loadB(vb + (size_t)48 * N_ + m0, N_);
    o0 = wmma_bf16(pA, vB0, o0);
    o1 = wmma_bf16(pA, vB1, o1);
    o2 = wmma_bf16(pA, vB2, o2);
    o3 = wmma_bf16(pA, vB3, o3);
  }

#pragma unroll
  for (int j = 0; j < 8; ++j) {
    float inv = 1.0f / rs[j];
    int row = n0 + j + 8 * rsel;
    size_t base = ((size_t)b * N_ + row) * C_ + (size_t)h * D_;
    attnO[base + 0  + col] = f2bf(o0[j] * inv);
    attnO[base + 16 + col] = f2bf(o1[j] * inv);
    attnO[base + 32 + col] = f2bf(o2[j] * inv);
    attnO[base + 48 + col] = f2bf(o3[j] * inv);
  }
}

}  // namespace

extern "C" void kernel_launch(void* const* d_in, const int* in_sizes, int n_in,
                              void* d_out, int out_size, void* d_ws, size_t ws_size,
                              hipStream_t stream) {
  const float* x    = (const float*)d_in[0];
  const float* wq   = (const float*)d_in[1];
  const float* wk   = (const float*)d_in[2];
  const float* wv   = (const float*)d_in[3];
  const float* bv   = (const float*)d_in[4];
  const float* cq_w = (const float*)d_in[5];
  const float* cq_b = (const float*)d_in[6];
  const float* ck_w = (const float*)d_in[7];
  const float* ck_b = (const float*)d_in[8];
  const float* cv_w = (const float*)d_in[9];
  const float* cv_b = (const float*)d_in[10];
  const float* wo   = (const float*)d_in[11];
  const float* bo   = (const float*)d_in[12];
  float* out = (float*)d_out;

  const size_t MN = (size_t)B_ * N_;          // 8192
  const size_t szXB  = MN * C_ * 2;           // bf16 x
  const size_t szWT  = (size_t)C_ * HD_ * 2;  // bf16 transposed weight
  const size_t szF32 = MN * HD_ * 4;          // fp32 q/k/v
  const size_t szBF  = MN * HD_ * 2;          // bf16 reordered tensors

  char* ws = (char*)d_ws;
  size_t off = 0;
  __bf16* xb  = (__bf16*)(ws + off); off += szXB;
  __bf16* wqT = (__bf16*)(ws + off); off += szWT;
  __bf16* wkT = (__bf16*)(ws + off); off += szWT;
  __bf16* wvT = (__bf16*)(ws + off); off += szWT;
  __bf16* woT = (__bf16*)(ws + off); off += szWT;
  float*  qf  = (float*)(ws + off);  off += szF32;
  float*  kf  = (float*)(ws + off);  off += szF32;
  float*  vf  = (float*)(ws + off);  off += szF32;
  __bf16* qc  = (__bf16*)(ws + off); off += szBF;
  __bf16* kc  = (__bf16*)(ws + off); off += szBF;
  __bf16* vT  = (__bf16*)(ws + off); off += szBF;
  __bf16* ao  = (__bf16*)(ws + off); off += szBF;

  // 1. convert inputs
  {
    int n = (int)(MN * C_);
    cvt_bf16_kernel<<<(n + 255) / 256, 256, 0, stream>>>(x, xb, n);
    int nw = C_ * HD_;
    int blk = (nw + 255) / 256;
    transpose_cvt_kernel<<<blk, 256, 0, stream>>>(wq, wqT, C_, HD_);
    transpose_cvt_kernel<<<blk, 256, 0, stream>>>(wk, wkT, C_, HD_);
    transpose_cvt_kernel<<<blk, 256, 0, stream>>>(wv, wvT, C_, HD_);
    transpose_cvt_kernel<<<blk, 256, 0, stream>>>(wo, woT, C_, C_);
  }

  // 2. Q/K/V projections (WMMA)
  {
    dim3 grid(MN / 32, HD_ / 64);
    gemm_bf16_kernel<<<grid, 32, 0, stream>>>(xb, wqT, nullptr, qf, (int)MN, HD_, C_);
    gemm_bf16_kernel<<<grid, 32, 0, stream>>>(xb, wkT, nullptr, kf, (int)MN, HD_, C_);
    gemm_bf16_kernel<<<grid, 32, 0, stream>>>(xb, wvT, bv,      vf, (int)MN, HD_, C_);
  }

  // 3. causal depthwise conv over batch + reorder to attention layouts
  {
    int n = N_ * HD_;
    conv_reorder_kernel<<<(n + 255) / 256, 256, 0, stream>>>(
        qf, kf, vf, cq_w, cq_b, ck_w, ck_b, cv_w, cv_b, qc, kc, vT);
  }

  // 4. attention (WMMA flash-style)
  {
    dim3 grid(N_ / 16, B_ * H_);
    attn_kernel<<<grid, 32, 0, stream>>>(qc, kc, vT, ao);
  }

  // 5. output projection (WMMA) -> d_out fp32
  {
    dim3 grid(MN / 32, C_ / 64);
    gemm_bf16_kernel<<<grid, 32, 0, stream>>>(ao, woT, bo, out, (int)MN, C_, C_);
  }
}
